// MoE_14285061226918
// MI455X (gfx1250) — matrix operand order repaired and checked by reading, not compiled, not run on hardware
//
#include <hip/hip_runtime.h>

// ---------------- problem constants ----------------
constexpr int BB = 4, SS = 2048, DD = 1024, HH = 2816, EE = 8;
constexpr int TT = BB * SS;            // 8192 tokens

typedef __attribute__((ext_vector_type(16))) __bf16 v16bf;
typedef __attribute__((ext_vector_type(8)))  __bf16 v8bf;
typedef __attribute__((ext_vector_type(8)))  float  v8f;

// round-to-nearest-even fp32 -> bf16 (as raw ushort)
__device__ __forceinline__ unsigned short f2bf(float f) {
    unsigned int u = __float_as_uint(f);
    unsigned int r = u + 0x7FFFu + ((u >> 16) & 1u);
    return (unsigned short)(r >> 16);
}

__device__ __forceinline__ v16bf mk16(v8bf lo, v8bf hi) {
    v16bf a;
#pragma unroll
    for (int i = 0; i < 8; i++) { a[i] = lo[i]; a[8 + i] = hi[i]; }
    return a;
}

// async global->LDS 16B copy (gfx1250 GLOBAL_LOAD_ASYNC_TO_LDS_B128, tracked by ASYNCcnt)
__device__ __forceinline__ void async_cp16(unsigned ldsOff, const void* gaddr) {
    asm volatile("global_load_async_to_lds_b128 %0, %1, off" :: "v"(ldsOff), "v"(gaddr) : "memory");
}
__device__ __forceinline__ void async_wait0() {     // all copies complete
    asm volatile("s_wait_asynccnt 0x0" ::: "memory");
}
__device__ __forceinline__ void async_wait4() {     // <=4 outstanding: previous slab (in-order) done
    asm volatile("s_wait_asynccnt 0x4" ::: "memory");
}

// ---------------- utility kernels ----------------
__global__ void zero_f32(float* p, int n) {
    int i = blockIdx.x * blockDim.x + threadIdx.x;
    if (i < n) p[i] = 0.0f;
}

__global__ void cvt_f32_bf16(const float* s, unsigned short* d, int n) {
    int i = blockIdx.x * blockDim.x + threadIdx.x;
    if (i < n) d[i] = f2bf(s[i]);
}

// [R][C] -> [C][R] per matrix (blockIdx.z), fp32 in, bf16 out, LDS-tiled
__global__ __launch_bounds__(256) void transpose_cvt(const float* __restrict__ src,
                                                     unsigned short* __restrict__ dst,
                                                     int R, int C) {
    __shared__ unsigned short tile[32][33];
    const float* s = src + (size_t)blockIdx.z * R * C;
    unsigned short* d = dst + (size_t)blockIdx.z * R * C;
    int c0 = blockIdx.x * 32, r0 = blockIdx.y * 32;
    for (int i = threadIdx.y; i < 32; i += 8) {
        int r = r0 + i, c = c0 + threadIdx.x;
        float v = (r < R && c < C) ? s[(size_t)r * C + c] : 0.0f;
        tile[i][threadIdx.x] = f2bf(v);
    }
    __syncthreads();
    for (int i = threadIdx.y; i < 32; i += 8) {
        int r = r0 + threadIdx.x, c = c0 + i;
        if (r < R && c < C) d[(size_t)c * R + r] = tile[threadIdx.x][i];
    }
}

// ---------------- routing: scores, top-2, softmax -> coef[T][E] ----------------
__global__ __launch_bounds__(256) void routing_kernel(const float* __restrict__ x,
                                                      const float* __restrict__ gate,
                                                      float* __restrict__ coef) {
    __shared__ float g[EE * DD];                 // 32 KB
    for (int i = threadIdx.x; i < EE * DD; i += 256) g[i] = gate[i];
    __syncthreads();
    int wave = threadIdx.x >> 5, lane = threadIdx.x & 31;
    int t = blockIdx.x * 8 + wave;               // one wave per token
    float acc[EE];
#pragma unroll
    for (int e = 0; e < EE; e++) acc[e] = 0.0f;
    for (int d = lane; d < DD; d += 32) {
        float xv = x[(size_t)t * DD + d];
#pragma unroll
        for (int e = 0; e < EE; e++) acc[e] += xv * g[e * DD + d];
    }
#pragma unroll
    for (int e = 0; e < EE; e++)
        for (int off = 16; off; off >>= 1) acc[e] += __shfl_xor(acc[e], off, 32);
    if (lane == 0) {
        int i1 = 0; float s1 = acc[0];
        for (int e = 1; e < EE; e++) if (acc[e] > s1) { s1 = acc[e]; i1 = e; }
        int i2 = (i1 == 0) ? 1 : 0; float s2 = acc[i2];
        for (int e = 0; e < EE; e++) if (e != i1 && acc[e] > s2) { s2 = acc[e]; i2 = e; }
        float e2 = __expf(s2 - s1);
        float inv = 1.0f / (1.0f + e2);
#pragma unroll
        for (int e = 0; e < EE; e++) coef[t * EE + e] = 0.0f;
        coef[t * EE + i1] = inv;
        coef[t * EE + i2] = e2 * inv;
    }
}

// deterministic per-expert token lists (no atomics => bitwise repeatable tiling)
__global__ void build_lists(const float* __restrict__ coef, int* __restrict__ idxlist,
                            int* __restrict__ counts) {
    int e = threadIdx.x;
    if (e >= EE) return;
    int c = 0;
    for (int t = 0; t < TT; t++)
        if (coef[t * EE + e] > 0.0f) { idxlist[e * TT + c] = t; c++; }
    counts[e] = c;
}

// Shared tiling constants for the expert GEMMs:
constexpr int KS  = 128;             // K-slab elements (1024 = 8*128, 2816 = 22*128)
constexpr int STR = KS + 8;          // LDS row stride: 272B = 68 dwords -> lane bank step 4, conflict-free
constexpr int BUFB = 64 * STR * 2;   // one A-slab buffer: 17,408 B

// ---------------- expert up-projection: h = silu(x@w1) * (x@w3) ----------------
// Block = 8 waves (256 thr), tile M=64 x N=128. Wave (mg=w>>2, ng=w&3):
// quadrant M rows [mg*32,+32), N cols [ng*32,+32); 8 WMMA accumulators.
// Gathered A rows async-staged in double-buffered LDS slabs (copy overlaps compute).
__global__ __launch_bounds__(256) void moe_up(const unsigned short* __restrict__ xb,
                                              const unsigned short* __restrict__ w1t,
                                              const unsigned short* __restrict__ w3t,
                                              const int* __restrict__ idxlist,
                                              const int* __restrict__ counts,
                                              unsigned short* __restrict__ hbuf, int e) {
    __shared__ __attribute__((aligned(16))) unsigned short shA[2 * 64 * STR];  // 34,816 B
    __shared__ int shTok[64];
    int cnt = counts[e];
    int rowbase = blockIdx.y * 64;
    if (rowbase >= cnt) return;
    int tid = threadIdx.x;
    if (tid < 64) {
        int r = rowbase + tid;
        int rc = (r < cnt) ? r : (cnt - 1);      // clamp ragged tail
        shTok[tid] = idxlist[e * TT + rc];
    }
    __syncthreads();

    int wave = tid >> 5, lane = tid & 31, half = lane >> 4, l16 = lane & 15;
    int mg = wave >> 2, ng = wave & 3;
    int nc0 = blockIdx.x * 128 + ng * 32 + l16;                // nt=0 column of this lane
    const unsigned short* b1r0 = w1t + ((size_t)e * HH + nc0) * DD;
    const unsigned short* b1r1 = b1r0 + (size_t)16 * DD;
    const unsigned short* b3r0 = w3t + ((size_t)e * HH + nc0) * DD;
    const unsigned short* b3r1 = b3r0 + (size_t)16 * DD;
    unsigned ldsBase = (unsigned)(unsigned long long)(uintptr_t)shA;  // low 32 bits = LDS offset

    // slab copier: 64 rows x 256B = 16 chunks/row, 1024 chunks, 4 per thread
    auto copy_slab = [&](int ks, unsigned bufOff) {
        for (int c = tid; c < 64 * 16; c += 256) {
            int row = c >> 4, cb = (c & 15) << 4;
            const unsigned short* g = xb + (size_t)shTok[row] * DD + ks + (cb >> 1);
            async_cp16(bufOff + (unsigned)(row * (STR * 2) + cb), g);
        }
    };

    v8f a1[2][2] = {}; v8f a2[2][2] = {};
    constexpr int NSLAB = DD / KS;               // 8
    copy_slab(0, ldsBase);
    for (int s = 0; s < NSLAB; s++) {
        unsigned cur = ldsBase + (unsigned)((s & 1) * BUFB);
        if (s + 1 < NSLAB) { copy_slab((s + 1) * KS, ldsBase + (unsigned)(((s + 1) & 1) * BUFB)); async_wait4(); }
        else async_wait0();
        __syncthreads();
        const unsigned short* bufp = shA + (size_t)(s & 1) * 64 * STR;
        for (int k = 0; k < KS; k += 32) {
            v16bf a[2];
#pragma unroll
            for (int mt = 0; mt < 2; mt++) {
                const unsigned short* ar = bufp + (size_t)(mg * 32 + mt * 16 + l16) * STR + k;
                a[mt] = mk16(*(const v8bf*)(ar + half * 16), *(const v8bf*)(ar + half * 16 + 8));
            }
            size_t kk = (size_t)(s * KS + k + half * 16);
            v16bf b10 = *(const v16bf*)(b1r0 + kk);
            v16bf b11 = *(const v16bf*)(b1r1 + kk);
            v16bf b30 = *(const v16bf*)(b3r0 + kk);
            v16bf b31 = *(const v16bf*)(b3r1 + kk);
#pragma unroll
            for (int mt = 0; mt < 2; mt++) {
                a1[mt][0] = __builtin_amdgcn_wmma_f32_16x16x32_bf16(false, a[mt], false, b10, (short)0, a1[mt][0], false, false);
                a1[mt][1] = __builtin_amdgcn_wmma_f32_16x16x32_bf16(false, a[mt], false, b11, (short)0, a1[mt][1], false, false);
                a2[mt][0] = __builtin_amdgcn_wmma_f32_16x16x32_bf16(false, a[mt], false, b30, (short)0, a2[mt][0], false, false);
                a2[mt][1] = __builtin_amdgcn_wmma_f32_16x16x32_bf16(false, a[mt], false, b31, (short)0, a2[mt][1], false, false);
            }
        }
        __syncthreads();   // all waves done with this slab before its buffer is re-filled
    }
    // epilogue: lane holds col nc0 + nt*16; VGPR i holds row mg*32 + mt*16 + i + half*8
#pragma unroll
    for (int mt = 0; mt < 2; mt++)
#pragma unroll
        for (int nt = 0; nt < 2; nt++)
#pragma unroll
            for (int i = 0; i < 8; i++) {
                int row = rowbase + mg * 32 + mt * 16 + i + half * 8;
                float gv = a1[mt][nt][i];
                float sil = gv / (1.0f + __expf(-gv));
                hbuf[(size_t)row * HH + nc0 + nt * 16] = f2bf(sil * a2[mt][nt][i]);
            }
}

// ---------------- expert down-projection + weighted scatter-add ----------------
// Block = 8 waves (256 thr), tile M=64 x N=256 over D. Wave (mg=w>>2, ng=w&3):
// M rows [mg*32,+32), N cols [ng*64,+64); 8 accumulators (M=32 x N=64).
__global__ __launch_bounds__(256) void moe_down(const unsigned short* __restrict__ hbuf,
                                                const unsigned short* __restrict__ w2t,
                                                const int* __restrict__ idxlist,
                                                const int* __restrict__ counts,
                                                const float* __restrict__ coef,
                                                float* __restrict__ out, int e) {
    __shared__ __attribute__((aligned(16))) unsigned short shA[2 * 64 * STR];  // 34,816 B
    int cnt = counts[e];
    int rowbase = blockIdx.y * 64;
    if (rowbase >= cnt) return;
    int tid = threadIdx.x;
    int wave = tid >> 5, lane = tid & 31, half = lane >> 4, l16 = lane & 15;
    int mg = wave >> 2, ng = wave & 3;
    int nc0 = blockIdx.x * 256 + ng * 64 + l16;
    const unsigned short* br0 = w2t + ((size_t)e * DD + nc0) * HH;
    unsigned ldsBase = (unsigned)(unsigned long long)(uintptr_t)shA;

    auto copy_slab = [&](int ks, unsigned bufOff) {
        for (int c = tid; c < 64 * 16; c += 256) {
            int row = c >> 4, cb = (c & 15) << 4;
            const unsigned short* g = hbuf + (size_t)(rowbase + row) * HH + ks + (cb >> 1);
            async_cp16(bufOff + (unsigned)(row * (STR * 2) + cb), g);
        }
    };

    v8f acc[2][4] = {};
    constexpr int NSLAB = HH / KS;               // 22
    copy_slab(0, ldsBase);
    for (int s = 0; s < NSLAB; s++) {
        if (s + 1 < NSLAB) { copy_slab((s + 1) * KS, ldsBase + (unsigned)(((s + 1) & 1) * BUFB)); async_wait4(); }
        else async_wait0();
        __syncthreads();
        const unsigned short* bufp = shA + (size_t)(s & 1) * 64 * STR;
        for (int k = 0; k < KS; k += 32) {
            v16bf a[2];
#pragma unroll
            for (int mt = 0; mt < 2; mt++) {
                const unsigned short* ar = bufp + (size_t)(mg * 32 + mt * 16 + l16) * STR + k;
                a[mt] = mk16(*(const v8bf*)(ar + half * 16), *(const v8bf*)(ar + half * 16 + 8));
            }
            size_t kk = (size_t)(s * KS + k + half * 16);
#pragma unroll
            for (int nt = 0; nt < 4; nt++) {
                v16bf b = *(const v16bf*)(br0 + (size_t)nt * 16 * HH + kk);
#pragma unroll
                for (int mt = 0; mt < 2; mt++)
                    acc[mt][nt] = __builtin_amdgcn_wmma_f32_16x16x32_bf16(false, a[mt], false, b, (short)0, acc[mt][nt], false, false);
            }
        }
        __syncthreads();
    }
#pragma unroll
    for (int mt = 0; mt < 2; mt++)
#pragma unroll
        for (int i = 0; i < 8; i++) {
            int row = rowbase + mg * 32 + mt * 16 + i + half * 8;
            if (row < cnt) {
                int tok = idxlist[e * TT + row];
                float cf = coef[tok * EE + e];
#pragma unroll
                for (int nt = 0; nt < 4; nt++)
                    out[(size_t)tok * DD + nc0 + nt * 16] += cf * acc[mt][nt][i];  // experts stream-ordered: race-free
            }
        }
}

// ---------------- launch ----------------
extern "C" void kernel_launch(void* const* d_in, const int* in_sizes, int n_in,
                              void* d_out, int out_size, void* d_ws, size_t ws_size,
                              hipStream_t stream) {
    const float* x    = (const float*)d_in[0];
    const float* gate = (const float*)d_in[1];
    const float* w1   = (const float*)d_in[2];
    const float* w3   = (const float*)d_in[3];
    const float* w2   = (const float*)d_in[4];
    float* out = (float*)d_out;

    char* ws = (char*)d_ws;
    size_t off = 0;
    auto alloc = [&](size_t bytes) { char* p = ws + off; off += (bytes + 255) & ~(size_t)255; return p; };
    unsigned short* xb   = (unsigned short*)alloc((size_t)TT * DD * 2);
    unsigned short* w1t  = (unsigned short*)alloc((size_t)EE * HH * DD * 2);
    unsigned short* w3t  = (unsigned short*)alloc((size_t)EE * HH * DD * 2);
    unsigned short* w2t  = (unsigned short*)alloc((size_t)EE * DD * HH * 2);
    unsigned short* hbuf = (unsigned short*)alloc((size_t)TT * HH * 2);
    float* coef          = (float*)alloc((size_t)TT * EE * 4);
    int*   idxlist       = (int*)alloc((size_t)EE * TT * 4);
    int*   counts        = (int*)alloc((size_t)EE * 4);

    zero_f32<<<(TT * DD + 1023) / 1024, 1024, 0, stream>>>(out, TT * DD);
    cvt_f32_bf16<<<(TT * DD + 1023) / 1024, 1024, 0, stream>>>(x, xb, TT * DD);
    dim3 tb(32, 8);
    transpose_cvt<<<dim3(HH / 32, DD / 32, EE), tb, 0, stream>>>(w1, w1t, DD, HH); // -> [E][H][D]
    transpose_cvt<<<dim3(HH / 32, DD / 32, EE), tb, 0, stream>>>(w3, w3t, DD, HH); // -> [E][H][D]
    transpose_cvt<<<dim3(DD / 32, HH / 32, EE), tb, 0, stream>>>(w2, w2t, HH, DD); // -> [E][D][H]

    routing_kernel<<<TT / 8, 256, 0, stream>>>(x, gate, coef);
    build_lists<<<1, EE, 0, stream>>>(coef, idxlist, counts);

    for (int e = 0; e < EE; e++) {
        moe_up  <<<dim3(HH / 128, TT / 64), 256, 0, stream>>>(xb, w1t, w3t, idxlist, counts, hbuf, e);
        moe_down<<<dim3(DD / 256, TT / 64), 256, 0, stream>>>(hbuf, w2t, idxlist, counts, coef, out, e);
    }
}
